// SSIMLoss_7859790151901
// MI455X (gfx1250) — matrix operand order repaired and checked by reading, not compiled, not run on hardware
//
#include <hip/hip_runtime.h>

// SSIM loss for 128 x (1,512,512) fp32 image pairs, 11-tap Gaussian window
// (sigma=1.5), VALID padding (502x502 map), out = 1 - mean(ssim_map).
//
// MI455X (gfx1250, wave32) strategy:
//  - One wave per 16x16 output tile; persistent waves walk 8 adjacent tiles
//    (same image row) reusing the shared 16-column input chunk.
//  - Separable blur as WMMA f16 matmuls vs a banded Gaussian matrix
//    A[m,k] = g[k-m] (f32 accumulate):
//      vertical:    Dv(16x16)  = A x B_cols(32x16)
//      horizontal:  O^T(16x16) = A x V^T(32x16)   <- same A both passes
//    V^T B-fragments are contiguous rows of the row-major LDS intermediate
//    -> 2 x ds_load_b128 per channel.
//  - 5 channels (x, y, x^2, y^2, xy) -> SSIM map -> masked sum.
//  - Deterministic reduction: per-block partials in d_ws, fixed-order
//    single-block finalize kernel.

typedef _Float16 v16h __attribute__((ext_vector_type(16)));
typedef _Float16 v8h  __attribute__((ext_vector_type(8)));
typedef float    v8f  __attribute__((ext_vector_type(8)));

#define IMG_H 512
#define IMG_W 512
#define OUT_HW 502            // 512 - 11 + 1
#define N_IMG 128             // 8 * 16
#define TILES_1D 32           // ceil(502/16)
#define WAVES_PER_BLOCK 8
#define TILES_PER_WAVE 8      // horizontally adjacent, same tile-row
#define N_TILES (N_IMG * TILES_1D * TILES_1D)       // 131072
#define N_WAVES (N_TILES / TILES_PER_WAVE)          // 16384
#define N_BLOCKS (N_WAVES / WAVES_PER_BLOCK)        // 2048

// Normalized Gaussian tap: g[d] = exp(-(d-5)^2/4.5)/sum, sum = 3.7592332
__device__ __forceinline__ float gauss_tap(int d) {
    if (d < 0 || d > 10) return 0.0f;
    float t = (float)(d - 5);
    return __expf(-t * t * (1.0f / 4.5f)) * 0.26601252f;
}

// Load a 16-row column chunk of x and y as f16 fragments (B-layout K half for
// this lane). Fast path: single base + immediate row offsets.
__device__ __forceinline__ void load_chunk(
    const float* __restrict__ xb, const float* __restrict__ yb,
    int rbase, int col, bool clamped, v16h& hx, v16h& hy)
{
    if (!clamped) {
        const float* px = xb + (size_t)rbase * IMG_W + col;
        const float* py = yb + (size_t)rbase * IMG_W + col;
#pragma unroll
        for (int k = 0; k < 16; ++k) {
            hx[k] = (_Float16)__builtin_nontemporal_load(px + k * IMG_W);
            hy[k] = (_Float16)__builtin_nontemporal_load(py + k * IMG_W);
        }
    } else {
        int c = col < (IMG_W - 1) ? col : (IMG_W - 1);
#pragma unroll
        for (int k = 0; k < 16; ++k) {
            int row = rbase + k;
            row = row < (IMG_H - 1) ? row : (IMG_H - 1);
            size_t off = (size_t)row * IMG_W + c;
            hx[k] = (_Float16)__builtin_nontemporal_load(xb + off);
            hy[k] = (_Float16)__builtin_nontemporal_load(yb + off);
        }
    }
}

__global__ __launch_bounds__(256) void ssim_tile_kernel(
    const float* __restrict__ x, const float* __restrict__ y,
    float* __restrict__ partials)
{
    __shared__ __align__(16) _Float16 lds_tr[WAVES_PER_BLOCK][16 * 32];
    __shared__ float wave_sum[WAVES_PER_BLOCK];

    const int wave = threadIdx.x >> 5;
    const int lane = threadIdx.x & 31;
    const int nlo  = lane & 15;
    const bool hi  = lane >= 16;

    // wave id -> (image, tile row, first tile col); 8 tiles share one row.
    const int w   = blockIdx.x * WAVES_PER_BLOCK + wave;
    const int img = w >> 7;
    const int tr  = (w >> 2) & (TILES_1D - 1);
    const int tc0 = (w & 3) * TILES_PER_WAVE;

    const int r0    = tr * 16;
    const int rbase = r0 + (hi ? 16 : 0);
    const bool row_clamp = (tr == TILES_1D - 1);
    const bool rows_full = (r0 + 15 < OUT_HW);

    // ---- Banded Gaussian weight fragment (A-layout, used by BOTH passes) ---
    // A[m,k] = g[k-m]; lanes hold M = lane&15, K half = hi ? 16 : 0;
    // VGPR j -> K = khalf+2j.
    const int khA = hi ? 16 : 0;
    v16h Aw;
#pragma unroll
    for (int j = 0; j < 8; ++j) {
        int kb = khA + 2 * j;
        Aw[2 * j]     = (_Float16)gauss_tap(kb - nlo);
        Aw[2 * j + 1] = (_Float16)gauss_tap(kb + 1 - nlo);
    }

    const float* __restrict__ xb = x + (size_t)img * (IMG_H * IMG_W);
    const float* __restrict__ yb = y + (size_t)img * (IMG_H * IMG_W);

    _Float16* ldsw = &lds_tr[wave][0];
    const int mbase = hi ? 8 : 0;
    const float c1 = 0.0001f;   // (0.01 * 1)^2
    const float c2 = 0.0009f;   // (0.03 * 1)^2
    float sum = 0.0f;

    v16h hx[2], hy[2];
    load_chunk(xb, yb, rbase, tc0 * 16 + nlo, row_clamp, hx[0], hy[0]);

    for (int t = 0; t < TILES_PER_WAVE; ++t) {
        const int tc = tc0 + t;
        const int c0 = tc * 16;
        // chunk1: cols c0+16 .. c0+31; needs col clamp only on last tile col.
        load_chunk(xb, yb, rbase, c0 + 16 + nlo,
                   row_clamp || (tc == TILES_1D - 1), hx[1], hy[1]);

        v8f res[5];
#pragma unroll
        for (int c = 0; c < 5; ++c) {
            // ---- vertical blur: two 16-col chunks -------------------------
            v8f vint[2];
#pragma unroll
            for (int chk = 0; chk < 2; ++chk) {
                v16h bfrag = (c == 0) ? hx[chk]
                           : (c == 1) ? hy[chk]
                           : (c == 2) ? hx[chk] * hx[chk]     // v_pk_mul_f16
                           : (c == 3) ? hy[chk] * hy[chk]
                                      : hx[chk] * hy[chk];
                v8f zero = {};
                vint[chk] = __builtin_amdgcn_wmma_f32_16x16x32_f16(
                    false, Aw, false, bfrag, (short)0, zero, false, false);
            }

            // ---- stage intermediate V (16 rows x 32 K, f16, row-major) ----
            // Dv layout: VGPR r -> row mbase+r, lane -> col chk*16+nlo.
#pragma unroll
            for (int chk = 0; chk < 2; ++chk)
#pragma unroll
                for (int r = 0; r < 8; ++r)
                    ldsw[(mbase + r) * 32 + chk * 16 + nlo] =
                        (_Float16)vint[chk][r];

            __asm__ volatile("s_wait_dscnt 0" ::: "memory");

            // ---- horizontal blur as O^T = A x V^T -------------------------
            // B-fragment of V^T: lane (N=nlo) reads row nlo of V, its K half
            // is contiguous -> two b128 LDS loads.
            const v8h* rp = (const v8h*)&ldsw[nlo * 32 + (hi ? 16 : 0)];
            v8h blo = rp[0];
            v8h bhi = rp[1];
            v16h bfragV = __builtin_shufflevector(
                blo, bhi, 0, 1, 2, 3, 4, 5, 6, 7, 8, 9, 10, 11, 12, 13, 14, 15);

            v8f zero = {};
            res[c] = __builtin_amdgcn_wmma_f32_16x16x32_f16(
                false, Aw, false, bfragV, (short)0, zero, false, false);

            __asm__ volatile("s_wait_dscnt 0" ::: "memory"); // WAR next chan
        }

        // ---- SSIM map (transposed tile) + masked accumulation -------------
        // O^T: out_row = r0 + nlo, out_col = c0 + mbase + r.
        const bool full = rows_full && (c0 + 15 < OUT_HW);
        const int orow = r0 + nlo;
#pragma unroll
        for (int r = 0; r < 8; ++r) {
            const int ocol = c0 + mbase + r;
            if (full || (orow < OUT_HW && ocol < OUT_HW)) {
                float mu1 = res[0][r], mu2 = res[1][r];
                float exx = res[2][r], eyy = res[3][r], exy = res[4][r];
                float mu1s = mu1 * mu1, mu2s = mu2 * mu2, mu12 = mu1 * mu2;
                float s1 = exx - mu1s, s2 = eyy - mu2s, s12 = exy - mu12;
                float cs = (2.0f * s12 + c2) / (s1 + s2 + c2);
                float lm = (2.0f * mu12 + c1) / (mu1s + mu2s + c1);
                sum += lm * cs;
            }
        }

        // slide the 16-column window
        hx[0] = hx[1];
        hy[0] = hy[1];
    }

    // ---- deterministic reduction ------------------------------------------
#pragma unroll
    for (int off = 16; off > 0; off >>= 1)
        sum += __shfl_xor(sum, off, 32);
    if (lane == 0) wave_sum[wave] = sum;
    __syncthreads();
    if (threadIdx.x == 0) {
        float b = 0.0f;
#pragma unroll
        for (int wv = 0; wv < WAVES_PER_BLOCK; ++wv) b += wave_sum[wv];
        partials[blockIdx.x] = b;
    }
}

__global__ __launch_bounds__(256) void ssim_finalize_kernel(
    const float* __restrict__ partials, float* __restrict__ out)
{
    __shared__ float sm[256];
    float s = 0.0f;
    for (int i = threadIdx.x; i < N_BLOCKS; i += 256) s += partials[i];
    sm[threadIdx.x] = s;
    __syncthreads();
    for (int stride = 128; stride > 0; stride >>= 1) {
        if ((int)threadIdx.x < stride) sm[threadIdx.x] += sm[threadIdx.x + stride];
        __syncthreads();
    }
    if (threadIdx.x == 0) {
        const float inv_n = 1.0f / ((float)N_IMG * (float)OUT_HW * (float)OUT_HW);
        out[0] = 1.0f - sm[0] * inv_n;
    }
}

extern "C" void kernel_launch(void* const* d_in, const int* in_sizes, int n_in,
                              void* d_out, int out_size, void* d_ws, size_t ws_size,
                              hipStream_t stream)
{
    (void)in_sizes; (void)n_in; (void)out_size; (void)ws_size;
    const float* seq1 = (const float*)d_in[0];
    const float* seq2 = (const float*)d_in[1];
    float* out      = (float*)d_out;
    float* partials = (float*)d_ws;  // N_BLOCKS floats (8 KB)

    ssim_tile_kernel<<<N_BLOCKS, 256, 0, stream>>>(seq1, seq2, partials);
    ssim_finalize_kernel<<<1, 256, 0, stream>>>(partials, out);
}
